// SelfAttention_47880295416383
// MI455X (gfx1250) — compile-verified, matrix operand order checked by reading, not run on hardware
//
#include <hip/hip_runtime.h>

// ---------------------------------------------------------------------------
// Causal multi-head self-attention forward for MI455X (gfx1250, wave32).
// Matmuls via v_wmma_f32_16x16x32_bf16; attention K/V tiles streamed into LDS
// by the Tensor Data Mover (tensor_load_to_lds, TENSORcnt-pipelined).
// ---------------------------------------------------------------------------

typedef __attribute__((ext_vector_type(16))) __bf16 v16bf;
typedef __attribute__((ext_vector_type(8)))  __bf16 v8bf;
typedef __attribute__((ext_vector_type(8)))  float  v8f;
typedef __attribute__((ext_vector_type(4)))  unsigned int u32x4;
typedef __attribute__((ext_vector_type(8)))  unsigned int u32x8;

#define BATCH 2
#define SEQ   2048
#define DIM   1024
#define HEADS 16
#define HD    64
#define MROWS (BATCH * SEQ)   // 4096

// ---------------- workspace layout (bytes) ---------------------------------
#define XB_OFF   ((size_t)0)                                  // x bf16  8 MB
#define WT_OFF   (XB_OFF + (size_t)MROWS * DIM * 2)           // 4x Wt   8 MB
#define Q_OFF    (WT_OFF + (size_t)4 * DIM * DIM * 2)         // Q bf16  8 MB
#define K_OFF    (Q_OFF + (size_t)MROWS * DIM * 2)            // K bf16  8 MB
#define VT_OFF   (K_OFF + (size_t)MROWS * DIM * 2)            // V^T     8 MB
#define CTX_OFF  (VT_OFF + (size_t)MROWS * DIM * 2)           // ctx     8 MB

// ---------------- WMMA helpers ---------------------------------------------
__device__ inline v8f wmma_bf16(v16bf a, v16bf b, v8f c) {
  // (neg_a, A, neg_b, B, c_mod, C, reuse_a, reuse_b)
  return __builtin_amdgcn_wmma_f32_16x16x32_bf16(false, a, false, b,
                                                 (short)0, c, false, false);
}

// Load a 16x32 bf16 fragment (A layout; also B layout when `base` points to a
// [N][K]-major matrix).  ISA 7.12.2: lanes 0-15 hold rows, K split 0-7/8-15
// between lane halves; two contiguous 16-byte loads per lane.
__device__ inline v16bf ldfrag(const __bf16* base, int ld, int k0, int lane) {
  const int r  = lane & 15;
  const int kb = (lane >> 4) << 3;          // 0 or 8
  const __bf16* p = base + (size_t)r * ld + (k0 + kb);
  v8bf lo = *reinterpret_cast<const v8bf*>(p);        // K = k0+kb .. +7
  v8bf hi = *reinterpret_cast<const v8bf*>(p + 16);   // K = k0+16+kb .. +7
  return __builtin_shufflevector(lo, hi, 0, 1, 2, 3, 4, 5, 6, 7,
                                         8, 9, 10, 11, 12, 13, 14, 15);
}

// A-fragment gather from ctx stored [b][h][s][d]: K dim = h*64+d.
__device__ inline v16bf ldfrag_ctx(const __bf16* ctx, int row0, int k0, int lane) {
  const int r  = lane & 15;
  const int m  = row0 + r;
  const int b  = m >> 11;              // / SEQ
  const int s  = m & (SEQ - 1);
  const int kb = (lane >> 4) << 3;
  const int k1 = k0 + kb;
  const int k2 = k0 + 16 + kb;
  const __bf16* p1 =
      ctx + (((size_t)(b * HEADS + (k1 >> 6)) * SEQ + s) * HD + (k1 & (HD - 1)));
  const __bf16* p2 =
      ctx + (((size_t)(b * HEADS + (k2 >> 6)) * SEQ + s) * HD + (k2 & (HD - 1)));
  v8bf lo = *reinterpret_cast<const v8bf*>(p1);
  v8bf hi = *reinterpret_cast<const v8bf*>(p2);
  return __builtin_shufflevector(lo, hi, 0, 1, 2, 3, 4, 5, 6, 7,
                                         8, 9, 10, 11, 12, 13, 14, 15);
}

// ---------------- Tensor Data Mover: 2-D tile -> LDS -----------------------
// Builds a D# per CDNA5 ISA (§8.3/8.4) and issues
// tensor_load_to_lds (VIMAGE).  Tracked with TENSORcnt.
// g0: [1:0]=count=1, [63:32]=lds_addr, [120:64]=global_addr, [127:126]=type(2)
// g1: [17:16]=data_size(1 => 2B), dim/tile fields per spec; groups 2/3 zero.
__device__ inline void tdm_load_2d(unsigned int lds_off, const void* gptr,
                                   unsigned int tile0, unsigned int tile1,
                                   unsigned long long stride0,
                                   unsigned int tdim0, unsigned int tdim1) {
  const unsigned long long ga = (unsigned long long)(size_t)gptr;
  u32x4 g0;
  g0[0] = 1u;                                   // count=1 (valid user D#)
  g0[1] = lds_off;                              // lds_addr (bytes)
  g0[2] = (unsigned int)ga;                     // global_addr[31:0]
  g0[3] = (unsigned int)(ga >> 32) | (2u << 30);// global_addr[56:32] | type=2
  u32x8 g1;
  g1[0] = 1u << 16;                             // data_size = 2 bytes
  g1[1] = (tdim0 & 0xFFFFu) << 16;              // tensor_dim0[15:0]
  g1[2] = (tdim0 >> 16) | ((tdim1 & 0xFFFFu) << 16);
  g1[3] = (tdim1 >> 16) | (tile0 << 16);        // tile_dim0
  g1[4] = tile1;                                // tile_dim1 (tile_dim2 = 0)
  g1[5] = (unsigned int)stride0;                // tensor_dim0_stride[31:0]
  g1[6] = (unsigned int)(stride0 >> 32);        // [47:32] | t1_stride[15:0]=0
  g1[7] = 0u;
  u32x4 gz;
  gz[0] = 0u; gz[1] = 0u; gz[2] = 0u; gz[3] = 0u;
  asm volatile("tensor_load_to_lds %0, %1, %2, %3"
               :: "s"(g0), "s"(g1), "s"(gz), "s"(gz)
               : "memory");
}

// ---------------- kernel 1: fp32 -> bf16 -----------------------------------
__global__ void cvt_bf16_kernel(const float* __restrict__ src,
                                __bf16* __restrict__ dst, int n) {
  int i = blockIdx.x * blockDim.x + threadIdx.x;
  if (i < n) dst[i] = (__bf16)src[i];
}

// ---------------- kernel 2: W[K][N] -> Wt[N][K] bf16 (LDS tiled) -----------
__global__ __launch_bounds__(256) void transpose_bf16_kernel(
    const float* __restrict__ W, __bf16* __restrict__ Wt) {
  __shared__ float tile[32][33];
  const int c  = blockIdx.x * 32 + threadIdx.x;  // source col (N)
  const int r0 = blockIdx.y * 32;                // source row base (K)
  for (int i = threadIdx.y; i < 32; i += 8)
    tile[i][threadIdx.x] = W[(size_t)(r0 + i) * DIM + c];
  __syncthreads();
  const int k  = r0 + threadIdx.x;
  const int n0 = blockIdx.x * 32;
  for (int i = threadIdx.y; i < 32; i += 8)
    Wt[(size_t)(n0 + i) * DIM + k] = (__bf16)tile[threadIdx.x][i];
}

// ---------------- kernel 3: fused QKV projection ---------------------------
__global__ __launch_bounds__(128) void qkv_wmma_kernel(
    const __bf16* __restrict__ xb,
    const __bf16* __restrict__ wqT, const __bf16* __restrict__ wkT,
    const __bf16* __restrict__ wvT,
    const float* __restrict__ bq, const float* __restrict__ bk,
    const float* __restrict__ bv,
    __bf16* __restrict__ qo, __bf16* __restrict__ ko,
    __bf16* __restrict__ vTo) {
  const int lane = threadIdx.x & 31;
  const int wave = threadIdx.x >> 5;
  const int row0 = blockIdx.x * 16;
  const int n0   = blockIdx.y * 256 + wave * 64;

  v8f aq[4] = {}, ak[4] = {}, av[4] = {};
  for (int k0 = 0; k0 < DIM; k0 += 32) {
    if (k0 + 32 < DIM)
      __builtin_prefetch(xb + (size_t)row0 * DIM + k0 + 32, 0, 0);
    v16bf a = ldfrag(xb + (size_t)row0 * DIM, DIM, k0, lane);
#pragma unroll
    for (int n = 0; n < 4; ++n) {
      const size_t wrow = (size_t)(n0 + 16 * n) * DIM;
      aq[n] = wmma_bf16(a, ldfrag(wqT + wrow, DIM, k0, lane), aq[n]);
      ak[n] = wmma_bf16(a, ldfrag(wkT + wrow, DIM, k0, lane), ak[n]);
      av[n] = wmma_bf16(a, ldfrag(wvT + wrow, DIM, k0, lane), av[n]);
    }
  }
  const int rb = (lane < 16) ? 0 : 8;   // C/D layout: M = vgpr + 8*(lane>=16)
  const int cl = lane & 15;             // N = lane & 15
#pragma unroll
  for (int n = 0; n < 4; ++n) {
#pragma unroll
    for (int r = 0; r < 8; ++r) {
      const int m   = row0 + rb + r;
      const int col = n0 + 16 * n + cl;
      const int b = m >> 11, s = m & (SEQ - 1);
      const int h = col >> 6, d = col & (HD - 1);
      const size_t bh = (size_t)b * HEADS + h;
      qo[(bh * SEQ + s) * HD + d]  = (__bf16)(aq[n][r] + bq[col]);
      ko[(bh * SEQ + s) * HD + d]  = (__bf16)(ak[n][r] + bk[col]);
      vTo[(bh * HD + d) * SEQ + s] = (__bf16)(av[n][r] + bv[col]);
    }
  }
}

// ---------------- kernel 4: causal flash attention (TDM-pipelined) ---------
// One wave owns a 16-row q-tile.  32-key K/V tiles are streamed into per-wave
// double-buffered LDS by the Tensor Data Mover, overlapped with WMMA compute.
__global__ __launch_bounds__(128) void attn_wmma_kernel(
    const __bf16* __restrict__ q, const __bf16* __restrict__ k,
    const __bf16* __restrict__ vT, __bf16* __restrict__ ctx) {
  const int lane = threadIdx.x & 31;
  const int wave = threadIdx.x >> 5;
  const int bh   = blockIdx.y;
  const int q0   = blockIdx.x * 64 + wave * 16;

  const __bf16* qh = q  + (size_t)bh * SEQ * HD;
  const __bf16* kh = k  + (size_t)bh * SEQ * HD;
  const __bf16* vh = vT + (size_t)bh * HD * SEQ;

  // Q tile 16x64 as two A-fragments (resident for whole loop)
  const v16bf qa0 = ldfrag(qh + (size_t)q0 * HD, HD, 0, lane);
  const v16bf qa1 = ldfrag(qh + (size_t)q0 * HD, HD, 32, lane);

  v8f acc[4] = {};
  float mrow[8], lrow[8];
#pragma unroll
  for (int r = 0; r < 8; ++r) { mrow[r] = -3.0e38f; lrow[r] = 0.0f; }

  const int rb = (lane < 16) ? 0 : 8;
  const int cl = lane & 15;
  const float scale = 0.125f;  // 1/sqrt(64)

  // per-wave LDS: double-buffered K (32x64) and V (64x32) tiles + P staging
  alignas(16) __shared__ __bf16 ktile[4][2][32 * HD];   // 32 KB
  alignas(16) __shared__ __bf16 vtile[4][2][HD * 32];   // 32 KB
  alignas(16) __shared__ __bf16 plds[4][16 * 32];       //  4 KB
  __bf16* pl = plds[wave];

  const unsigned int koff0 =
      (unsigned int)__builtin_amdgcn_readfirstlane(
          (int)(unsigned int)(size_t)&ktile[wave][0][0]);
  const unsigned int voff0 =
      (unsigned int)__builtin_amdgcn_readfirstlane(
          (int)(unsigned int)(size_t)&vtile[wave][0][0]);

  const int ntiles = (q0 >> 5) + 1;  // key tiles 0..q0 step 32

  // issue tile 0 (K: 32 rows x 64 cols; V^T: 64 rows x 32 cols)
  tdm_load_2d(koff0, kh, HD, 32, HD, HD, SEQ);
  tdm_load_2d(voff0, vh, 32, HD, SEQ, SEQ, HD);

  for (int t = 0; t < ntiles; ++t) {
    const int kt  = t << 5;
    const int buf = t & 1;
    if (t + 1 < ntiles) {
      // WAR guard: make sure old ds_loads from buf^1 have drained, then
      // prefetch tile t+1 into the other buffer and wait for tile t only.
      asm volatile("s_wait_dscnt 0x0" ::: "memory");
      const int kn = kt + 32;
      tdm_load_2d(koff0 + (buf ^ 1) * (32 * HD * 2), kh + (size_t)kn * HD,
                  HD, 32, HD, HD, SEQ);
      tdm_load_2d(voff0 + (buf ^ 1) * (HD * 32 * 2), vh + kn,
                  32, HD, SEQ, SEQ, HD);
      __builtin_amdgcn_s_wait_tensorcnt(2);  // tile t complete (in-order TDM)
    } else {
      __builtin_amdgcn_s_wait_tensorcnt(0);
    }
    const __bf16* kl = &ktile[wave][buf][0];
    const __bf16* vl = &vtile[wave][buf][0];

    // ---- scores: S = Q * K^T (two 16-key halves, hd=64 => 2 WMMAs each) ----
    v8f z = {};
    v8f s0 = wmma_bf16(qa0, ldfrag(kl, HD, 0, lane), z);
    s0 = wmma_bf16(qa1, ldfrag(kl, HD, 32, lane), s0);
    v8f s1 = wmma_bf16(qa0, ldfrag(kl + 16 * HD, HD, 0, lane), z);
    s1 = wmma_bf16(qa1, ldfrag(kl + 16 * HD, HD, 32, lane), s1);

    // ---- online softmax in C-fragment layout ------------------------------
    float fac[8];
#pragma unroll
    for (int r = 0; r < 8; ++r) {
      const int qrow = q0 + rb + r;
      float x0 = s0[r] * scale;
      float x1 = s1[r] * scale;
      if (kt + cl > qrow)      x0 = -1.0e30f;   // causal mask
      if (kt + 16 + cl > qrow) x1 = -1.0e30f;
      float rm = fmaxf(x0, x1);
      rm = fmaxf(rm, __shfl_xor(rm, 1));
      rm = fmaxf(rm, __shfl_xor(rm, 2));
      rm = fmaxf(rm, __shfl_xor(rm, 4));
      rm = fmaxf(rm, __shfl_xor(rm, 8));
      const float mnew = fmaxf(mrow[r], rm);
      const float p0 = __expf(x0 - mnew);
      const float p1 = __expf(x1 - mnew);
      float rs = p0 + p1;
      rs += __shfl_xor(rs, 1);
      rs += __shfl_xor(rs, 2);
      rs += __shfl_xor(rs, 4);
      rs += __shfl_xor(rs, 8);
      const float f = __expf(mrow[r] - mnew);
      lrow[r] = lrow[r] * f + rs;
      mrow[r] = mnew;
      fac[r]  = f;
      pl[(rb + r) * 32 + cl]      = (__bf16)p0;
      pl[(rb + r) * 32 + 16 + cl] = (__bf16)p1;
    }
#pragma unroll
    for (int n = 0; n < 4; ++n)
#pragma unroll
      for (int r = 0; r < 8; ++r) acc[n][r] *= fac[r];

    // ---- ctx += P * V -----------------------------------------------------
    v16bf pa = ldfrag(pl, 32, 0, lane);
#pragma unroll
    for (int n = 0; n < 4; ++n) {
      v16bf vb = ldfrag(vl + (size_t)(n * 16) * 32, 32, 0, lane);
      acc[n] = wmma_bf16(pa, vb, acc[n]);
    }
  }

  // ---- normalize and store ctx [b][h][s][d] -------------------------------
#pragma unroll
  for (int r = 0; r < 8; ++r) {
    const float inv = 1.0f / lrow[r];
    const int m = q0 + rb + r;
#pragma unroll
    for (int n = 0; n < 4; ++n) {
      const int d = n * 16 + cl;
      ctx[((size_t)bh * SEQ + m) * HD + d] = (__bf16)(acc[n][r] * inv);
    }
  }
}

// ---------------- kernel 5: output projection (fp32 out) -------------------
__global__ __launch_bounds__(128) void outproj_wmma_kernel(
    const __bf16* __restrict__ ctx, const __bf16* __restrict__ woT,
    const float* __restrict__ bo, float* __restrict__ out) {
  const int lane = threadIdx.x & 31;
  const int wave = threadIdx.x >> 5;
  const int row0 = blockIdx.x * 16;
  const int n0   = blockIdx.y * 256 + wave * 64;

  v8f acc[4] = {};
  for (int k0 = 0; k0 < DIM; k0 += 32) {
    v16bf a = ldfrag_ctx(ctx, row0, k0, lane);
#pragma unroll
    for (int n = 0; n < 4; ++n) {
      v16bf b = ldfrag(woT + (size_t)(n0 + 16 * n) * DIM, DIM, k0, lane);
      acc[n] = wmma_bf16(a, b, acc[n]);
    }
  }
  const int rb = (lane < 16) ? 0 : 8;
  const int cl = lane & 15;
#pragma unroll
  for (int n = 0; n < 4; ++n)
#pragma unroll
    for (int r = 0; r < 8; ++r) {
      const int m   = row0 + rb + r;
      const int col = n0 + 16 * n + cl;
      out[(size_t)m * DIM + col] = acc[n][r] + bo[col];
    }
}

// ---------------------------------------------------------------------------
extern "C" void kernel_launch(void* const* d_in, const int* in_sizes, int n_in,
                              void* d_out, int out_size, void* d_ws,
                              size_t ws_size, hipStream_t stream) {
  const float* x  = (const float*)d_in[0];
  const float* Wq = (const float*)d_in[1];
  const float* bq = (const float*)d_in[2];
  const float* Wk = (const float*)d_in[3];
  const float* bk = (const float*)d_in[4];
  const float* Wv = (const float*)d_in[5];
  const float* bv = (const float*)d_in[6];
  const float* Wo = (const float*)d_in[7];
  const float* bo = (const float*)d_in[8];
  float* out = (float*)d_out;

  char* ws = (char*)d_ws;
  __bf16* xb  = (__bf16*)(ws + XB_OFF);
  __bf16* wqT = (__bf16*)(ws + WT_OFF);
  __bf16* wkT = wqT + (size_t)DIM * DIM;
  __bf16* wvT = wkT + (size_t)DIM * DIM;
  __bf16* woT = wvT + (size_t)DIM * DIM;
  __bf16* qb  = (__bf16*)(ws + Q_OFF);
  __bf16* kb  = (__bf16*)(ws + K_OFF);
  __bf16* vTb = (__bf16*)(ws + VT_OFF);
  __bf16* ctx = (__bf16*)(ws + CTX_OFF);

  {
    const int n = MROWS * DIM;
    cvt_bf16_kernel<<<(n + 255) / 256, 256, 0, stream>>>(x, xb, n);
  }
  {
    dim3 g(DIM / 32, DIM / 32), b(32, 8);
    transpose_bf16_kernel<<<g, b, 0, stream>>>(Wq, wqT);
    transpose_bf16_kernel<<<g, b, 0, stream>>>(Wk, wkT);
    transpose_bf16_kernel<<<g, b, 0, stream>>>(Wv, wvT);
    transpose_bf16_kernel<<<g, b, 0, stream>>>(Wo, woT);
  }
  qkv_wmma_kernel<<<dim3(MROWS / 16, DIM / 256), 128, 0, stream>>>(
      xb, wqT, wkT, wvT, bq, bk, bv, qb, kb, vTb);
  attn_wmma_kernel<<<dim3(SEQ / 64, BATCH * HEADS), 128, 0, stream>>>(
      qb, kb, vTb, ctx);
  outproj_wmma_kernel<<<dim3(MROWS / 16, DIM / 256), 128, 0, stream>>>(
      ctx, woT, bo, out);
}
